// ProductKeyMemory_42606075576709
// MI455X (gfx1250) — matrix operand order repaired and checked by reading, not compiled, not run on hardware
//
#include <hip/hip_runtime.h>
#include <hip/hip_bf16.h>
#include <math.h>

// CDNA5 WMMA / TDM vector types
typedef __attribute__((ext_vector_type(16))) __bf16 v16bf;
typedef __attribute__((ext_vector_type(8)))  float  v8f;
typedef unsigned int u32x4 __attribute__((ext_vector_type(4)));
typedef int          i32x8 __attribute__((ext_vector_type(8)));
typedef int          i32x4 __attribute__((ext_vector_type(4)));

// Problem constants (B=4, T=4096, D=1024, SUB=512, VDIM=128, TOPK=8)
#define DFULL   1024
#define HALFD   512
#define SUBK    512
#define VDIM    128
#define RPW     64          // rows per workgroup (4 waves x 16 rows)
#define NWAVE   4
#define NT      (SUBK/16)   // 32 key tiles
#define BS      520         // padded bf16 LDS stride for B tile (TDM pad: 4 dwords / 256 dwords)
#define SS      516         // padded f32 score stride

#if __has_builtin(__builtin_amdgcn_sched_barrier)
#define SCHED_BARRIER() __builtin_amdgcn_sched_barrier(0)
#else
#define SCHED_BARRIER()
#endif

// ---------------- codebook f32 -> bf16 conversion (once per launch) ----------------
__global__ __launch_bounds__(256) void pkm_cvt_kernel(const float* __restrict__ a,
                                                      const float* __restrict__ b,
                                                      __bf16* __restrict__ oa,
                                                      __bf16* __restrict__ ob, int n) {
    int i = blockIdx.x * 256 + threadIdx.x;
    if (i < n) {
        oa[i] = (__bf16)a[i];
        ob[i] = (__bf16)b[i];
    }
}

// ---------------- query f32 -> bf16 conversion (vectorized, once per launch) -------
__global__ __launch_bounds__(256) void pkm_qcvt_kernel(const float* __restrict__ q,
                                                       __bf16* __restrict__ oq, int n4) {
    int i = blockIdx.x * 256 + threadIdx.x;
    if (i < n4) {
        float4 f = reinterpret_cast<const float4*>(q)[i];
        union { __bf16 b[4]; uint2 u; } x;
        x.b[0] = (__bf16)f.x; x.b[1] = (__bf16)f.y;
        x.b[2] = (__bf16)f.z; x.b[3] = (__bf16)f.w;
        reinterpret_cast<uint2*>(oq)[i] = x.u;
    }
}

#if __has_builtin(__builtin_amdgcn_tensor_load_to_lds)
// TDM D#: 2D tile 512(K) x 16(keys) bf16, tensor stride 512 elements,
// LDS pad 4 dwords every 256 dwords -> padded row stride 1040B (= BS bf16)
__device__ __forceinline__ void tdm_load_tile(const __bf16* cb, int t, __bf16* dst) {
    unsigned long long ga =
        (unsigned long long)(size_t)cb + (unsigned long long)t * 16u * HALFD * 2u;
    unsigned int ldsa = (unsigned int)(size_t)dst;
    u32x4 g0;
    g0[0] = 1u;                                   // count=1, user mode
    g0[1] = ldsa;                                 // lds_addr
    g0[2] = (unsigned int)(ga & 0xffffffffull);   // global_addr[31:0]
    g0[3] = (unsigned int)((ga >> 32) & 0x1ffffffull) | (2u << 30); // addr[56:32], type=2
    i32x8 g1;
    g1[0] = (int)((1u << 16) | (1u << 20) | (7u << 22) | (3u << 25));
                     // data_size=2B, pad_enable, pad_interval=256dw, pad_amount=4dw
    g1[1] = (int)(512u << 16);   // tensor_dim0 = 512
    g1[2] = (int)(512u << 16);   // tensor_dim1 = 512
    g1[3] = (int)(512u << 16);   // tile_dim0   = 512
    g1[4] = 16;                  // tile_dim1   = 16
    g1[5] = 512;                 // tensor_dim0_stride = 512
    g1[6] = 0;
    g1[7] = 0;
    i32x4 gz4 = {0, 0, 0, 0};
    i32x8 gz8 = {0, 0, 0, 0, 0, 0, 0, 0};
    __builtin_amdgcn_tensor_load_to_lds(g0, g1, gz4, gz4, gz8, 0);
}
#define HAVE_TDM 1
#else
#define HAVE_TDM 0
#endif

union BFrag { uint4 u[2]; v16bf v; };
__device__ __forceinline__ void load_bfrag(BFrag& d, const __bf16* p) {
    d.u[0] = *reinterpret_cast<const uint4*>(p);
    d.u[1] = *reinterpret_cast<const uint4*>(p + 16);
}

// ---------------- main product-key-memory kernel ----------------
template <bool QBF16>
__global__ __launch_bounds__(128) void pkm_kernel(const float* __restrict__ query,
                                                  const __bf16* __restrict__ qbf,
                                                  const __bf16* __restrict__ cb1b,
                                                  const __bf16* __restrict__ cb2b,
                                                  const float* __restrict__ values,
                                                  float* __restrict__ out, int nrows) {
    // LDS: per-wave f32 scores, double-buffered B tile, weights  (~170 KB total)
    __shared__ float  sc[NWAVE * 16 * SS];     // 132096 B
    __shared__ __bf16 Bsh[2][16 * BS];         //  33280 B (TDM-padded row stride 1040B)
    __shared__ float  wsel[2][RPW][8];         //   4096 B
    __shared__ int    isel[2][RPW][8];         //   4096 B

    const int tid  = threadIdx.x;
    const int wave = tid >> 5;
    const int lane = tid & 31;
    const int row0 = blockIdx.x * RPW;
    if (row0 >= nrows) return;

    const int m    = lane & 15;   // A row / B key-in-tile / D col
    const int hsel = lane >> 4;   // K-half selector per WMMA lane layout
    float* swave = sc + wave * 16 * SS;

    for (int half = 0; half < 2; ++half) {
        const __bf16* cb = half ? cb2b : cb1b;

        // ---- load this wave's 16x512 A tile into bf16 registers ----
        v16bf Areg[16];
        if (QBF16) {
            // direct bf16 loads: 32 x b128 straight into Areg, no temporaries
            const __bf16* qb = qbf + (size_t)(row0 + wave * 16 + m) * DFULL + half * HALFD;
            #pragma unroll
            for (int kc = 0; kc < 16; ++kc) {
                union { uint4 u[2]; v16bf v; } a;
                const __bf16* p = qb + kc * 32 + hsel * 8;
                a.u[0] = *reinterpret_cast<const uint4*>(p);
                a.u[1] = *reinterpret_cast<const uint4*>(p + 16);
                Areg[kc] = a.v;
            }
        } else {
            // f32 path with conversion; throttle lookahead to cap register pressure
            const float* qb = query + (size_t)(row0 + wave * 16 + m) * DFULL + half * HALFD;
            #pragma unroll
            for (int kc = 0; kc < 16; ++kc) {
                const float* p = qb + kc * 32 + hsel * 8;
                float4 f0 = *reinterpret_cast<const float4*>(p);
                float4 f1 = *reinterpret_cast<const float4*>(p + 4);
                float4 f2 = *reinterpret_cast<const float4*>(p + 16);
                float4 f3 = *reinterpret_cast<const float4*>(p + 20);
                v16bf a = {(__bf16)f0.x, (__bf16)f0.y, (__bf16)f0.z, (__bf16)f0.w,
                           (__bf16)f1.x, (__bf16)f1.y, (__bf16)f1.z, (__bf16)f1.w,
                           (__bf16)f2.x, (__bf16)f2.y, (__bf16)f2.z, (__bf16)f2.w,
                           (__bf16)f3.x, (__bf16)f3.y, (__bf16)f3.z, (__bf16)f3.w};
                Areg[kc] = a;
                if ((kc & 3) == 3) SCHED_BARRIER();
            }
        }

#if HAVE_TDM
        if (wave == 0) tdm_load_tile(cb, 0, &Bsh[0][0]);   // prologue: tile 0 in flight
#endif

        // ---- 32 key tiles; B staged by TDM double-buffer, shared by all 4 waves ----
        for (int t = 0; t < NT; ++t) {
#if HAVE_TDM
            if (wave == 0) {
                if (t + 1 < NT) {
                    tdm_load_tile(cb, t + 1, &Bsh[(t + 1) & 1][0]);  // prefetch next tile
                    __builtin_amdgcn_s_wait_tensorcnt(1);            // tile t landed
                } else {
                    __builtin_amdgcn_s_wait_tensorcnt(0);
                }
            }
#else
            {   // fallback: cooperative load, 128 threads x 64B
                const int r = tid >> 3;              // 16 rows
                const int c = (tid & 7) * 32;        // 4 x 16B chunks per thread (col in bf16)
                const uint4* gp = reinterpret_cast<const uint4*>(cb + (size_t)(t * 16 + r) * HALFD + c);
                uint4* lp = reinterpret_cast<uint4*>(&Bsh[t & 1][r * BS + c]);
                lp[0] = gp[0]; lp[1] = gp[1]; lp[2] = gp[2]; lp[3] = gp[3];
            }
#endif
            __syncthreads();   // B tile t visible to all waves

            // software-pipelined B fragments: 3 rotating buffers, 2 load-pairs in flight
            const __bf16* bp0 = &Bsh[t & 1][m * BS + hsel * 8];
            BFrag bb[3];
            load_bfrag(bb[0], bp0);
            load_bfrag(bb[1], bp0 + 32);
            v8f acc = {};
            #pragma unroll
            for (int kc = 0; kc < 16; ++kc) {
                if (kc + 2 < 16) load_bfrag(bb[(kc + 2) % 3], bp0 + (kc + 2) * 32);
                SCHED_BARRIER();   // keep prefetch above the WMMA (forces live buffers)
                acc = __builtin_amdgcn_wmma_f32_16x16x32_bf16(
                        false, Areg[kc], false, bb[kc % 3].v, (short)0, acc, false, false);
            }
            // D layout: VGPR r -> row (hsel*8 + r), col = t*16 + (lane&15)
            const int colb = t * 16 + m;
            #pragma unroll
            for (int r = 0; r < 8; ++r)
                swave[(hsel * 8 + r) * SS + colb] = acc[r];

            __syncthreads();   // all waves done with Bsh[t&1] before TDM t+2 reuses it
        }

        // ---- top-8 + softmax: each wave processes its own 16 rows ----
        for (int rr = 0; rr < 16; ++rr) {
            float cand[16];
            #pragma unroll
            for (int c = 0; c < 16; ++c) cand[c] = swave[rr * SS + lane + 32 * c];
            float tv[8]; int ti[8];
            #pragma unroll
            for (int k = 0; k < 8; ++k) {
                float bv = -3.4e38f; int bi = 0x7fffffff;
                #pragma unroll
                for (int c = 0; c < 16; ++c)
                    if (cand[c] > bv) { bv = cand[c]; bi = lane + 32 * c; }
                #pragma unroll
                for (int off = 16; off > 0; off >>= 1) {
                    float ov = __shfl_xor(bv, off, 32);
                    int   oi = __shfl_xor(bi, off, 32);
                    if (ov > bv || (ov == bv && oi < bi)) { bv = ov; bi = oi; }
                }
                tv[k] = bv; ti[k] = bi;
                if (lane == (bi & 31)) cand[bi >> 5] = -3.4e38f;
            }
            if (lane == 0) {
                const float mx = tv[0];
                float w[8], s = 0.f;
                #pragma unroll
                for (int k = 0; k < 8; ++k) { w[k] = __expf(tv[k] - mx); s += w[k]; }
                const float inv = 1.f / s;
                const int row = wave * 16 + rr;
                #pragma unroll
                for (int k = 0; k < 8; ++k) {
                    wsel[half][row][k] = w[k] * inv;
                    isel[half][row][k] = ti[k];
                }
            }
        }
        __syncthreads();
    }

    // ---- product gather + weighted sum (values L2-resident), 16 rows per wave ----
    for (int rr = 0; rr < 16; ++rr) {
        const int row = wave * 16 + rr;
        float w1[8], w2[8]; int i1[8], i2[8];
        #pragma unroll
        for (int k = 0; k < 8; ++k) {
            w1[k] = wsel[0][row][k]; i1[k] = isel[0][row][k];
            w2[k] = wsel[1][row][k]; i2[k] = isel[1][row][k];
        }
        float4 acc = make_float4(0.f, 0.f, 0.f, 0.f);
        #pragma unroll 4
        for (int i = 0; i < 8; ++i) {      // 4x8 = 32 float4 loads in flight max
            const int   b1 = i1[i] * SUBK;
            const float wi = w1[i];
            #pragma unroll
            for (int j = 0; j < 8; ++j) {
                const float4 v = *(reinterpret_cast<const float4*>(
                                       values + (size_t)(b1 + i2[j]) * VDIM) + lane);
                const float w = wi * w2[j];
                acc.x += w * v.x; acc.y += w * v.y; acc.z += w * v.z; acc.w += w * v.w;
            }
        }
        *(reinterpret_cast<float4*>(out + (size_t)(row0 + row) * VDIM) + lane) = acc;
    }
}

extern "C" void kernel_launch(void* const* d_in, const int* in_sizes, int n_in,
                              void* d_out, int out_size, void* d_ws, size_t ws_size,
                              hipStream_t stream) {
    const float* query  = (const float*)d_in[0];
    const float* cb1    = (const float*)d_in[1];
    const float* cb2    = (const float*)d_in[2];
    const float* values = (const float*)d_in[3];
    float* out = (float*)d_out;

    const int ncb   = in_sizes[1];               // 512*512
    const int nrows = in_sizes[0] / DFULL;       // B*T = 16384
    const int grid  = (nrows + RPW - 1) / RPW;   // 256 workgroups

    __bf16* cb1b = (__bf16*)d_ws;
    __bf16* cb2b = cb1b + ncb;
    __bf16* qbf  = cb2b + ncb;

    pkm_cvt_kernel<<<(ncb + 255) / 256, 256, 0, stream>>>(cb1, cb2, cb1b, cb2b, ncb);

    const size_t need = (size_t)2 * ncb * sizeof(__bf16)
                      + (size_t)nrows * DFULL * sizeof(__bf16);
    if (ws_size >= need) {
        const int n4 = nrows * DFULL / 4;
        pkm_qcvt_kernel<<<(n4 + 255) / 256, 256, 0, stream>>>(query, qbf, n4);
        pkm_kernel<true><<<grid, 128, 0, stream>>>(query, qbf, cb1b, cb2b, values, out, nrows);
    } else {
        pkm_kernel<false><<<grid, 128, 0, stream>>>(query, nullptr, cb1b, cb2b, values, out, nrows);
    }
}
